// FeatureExtraction_82179904242018
// MI455X (gfx1250) — matrix operand order repaired and checked by reading, not compiled, not run on hardware
//
#include <hip/hip_runtime.h>
#include <math.h>

// ---------------------------------------------------------------------------
// DGCNN feature extraction for MI455X (gfx1250, wave32, WMMA).
// EdgeConv is refactored: relu([xi, xj-xi]W + b) == relu(xi(Wa-Wb)+b + xj*Wb)
// -> two dense GEMMs (WMMA f16, f32 acc) + gather/max epilogue.
// ---------------------------------------------------------------------------

typedef __attribute__((ext_vector_type(16))) _Float16 v16h;
typedef __attribute__((ext_vector_type(8)))  _Float16 h8;
typedef __attribute__((ext_vector_type(8)))  float    v8f;

#define BATCH 4
#define NPTS  4096
#define MTOT  (BATCH * NPTS)   // 16384
#define KSEL  33               // K_NN + 1 (self included, masked later)

// ---------------- workspace arena (all offsets 256B aligned) ----------------
constexpr size_t OFF_IDX = 0;                                   // int  [16384 x 33]
constexpr size_t OFF_XC  = OFF_IDX + (size_t)MTOT * KSEL * 4;   // f32  [16384 x 208] (x1|x2|x3)
constexpr size_t OFF_XH  = OFF_XC  + (size_t)MTOT * 208 * 4;    // f16  [16384 x <=512] gemm input / x4h
constexpr size_t OFF_UH  = OFF_XH  + (size_t)MTOT * 512 * 2;    // f16  [16384 x <=512] u / h1
constexpr size_t OFF_VH  = OFF_UH  + (size_t)MTOT * 512 * 2;    // f16  [16384 x <=512] v / h2
constexpr size_t OFF_WU  = OFF_VH  + (size_t)MTOT * 512 * 2;    // f16  weights (Wa-Wb)^T, Kpad
constexpr size_t OFF_WV  = OFF_WU  + 512ull * 256 * 2;          // f16  Wb^T
constexpr size_t OFF_WL1 = OFF_WV  + 512ull * 256 * 2;          // f16  Wl1^T
constexpr size_t OFF_WL2 = OFF_WL1 + 512ull * 256 * 2;          // f16  Wl2^T

// ---------------------------------------------------------------------------
// kNN: one workgroup per query point. Distances in LDS, 33x iterative argmin.
// src is row-major [B*N x ld]; features at columns [col0, col0+d).
// ---------------------------------------------------------------------------
__global__ __launch_bounds__(256) void knn_kernel(const float* __restrict__ src,
                                                  int ld, int col0, int d,
                                                  int* __restrict__ idxOut) {
    __shared__ float q[160];
    __shared__ float dist[NPTS];
    __shared__ float rmin[256];
    __shared__ int   rmidx[256];

    const int p = blockIdx.x;          // 0..16383
    const int i = p & (NPTS - 1);
    const int b = p >> 12;
    const int tid = threadIdx.x;
    const float* Xb = src + (size_t)b * NPTS * ld + col0;

    if (tid < d) q[tid] = Xb[(size_t)i * ld + tid];
    __syncthreads();

    for (int j = tid; j < NPTS; j += 256) {
        const float* xj = Xb + (size_t)j * ld;
        float acc = 0.f;
        for (int c = 0; c < d; ++c) { float t = xj[c] - q[c]; acc += t * t; }
        dist[j] = acc;
    }
    __syncthreads();

    for (int k = 0; k < KSEL; ++k) {
        float m = 3.0e38f; int mi = 0;
        for (int j = tid; j < NPTS; j += 256) {
            float dj = dist[j];
            if (dj < m) { m = dj; mi = j; }
        }
        rmin[tid] = m; rmidx[tid] = mi;
        __syncthreads();
        for (int s = 128; s > 0; s >>= 1) {
            if (tid < s) {
                float om = rmin[tid + s]; int oi = rmidx[tid + s];
                if (om < rmin[tid] || (om == rmin[tid] && oi < rmidx[tid])) {
                    rmin[tid] = om; rmidx[tid] = oi;
                }
            }
            __syncthreads();
        }
        if (tid == 0) {
            int win = rmidx[0];
            idxOut[(size_t)p * KSEL + k] = win;
            dist[win] = 3.0e38f;
        }
        __syncthreads();
    }
}

// ---------------------------------------------------------------------------
// Weight prep: f32 [ (split? 2C : C) x Cout ] -> f16 transposed [Cout x Kpad].
// split: WuT = (Wa - Wb)^T, WvT = Wb^T.  !split: WuT = W^T. K rows >= C zeroed.
// ---------------------------------------------------------------------------
__global__ __launch_bounds__(256) void prep_weight(const float* __restrict__ W,
                                                   int C, int Cout, int Kpad,
                                                   _Float16* __restrict__ WuT,
                                                   _Float16* __restrict__ WvT,
                                                   int split) {
    int t = blockIdx.x * blockDim.x + threadIdx.x;
    if (t >= Cout * Kpad) return;
    int n = t / Kpad, k = t % Kpad;
    if (split) {
        float wa = (k < C) ? W[(size_t)k * Cout + n] : 0.f;
        float wb = (k < C) ? W[(size_t)(C + k) * Cout + n] : 0.f;
        WuT[t] = (_Float16)(wa - wb);
        WvT[t] = (_Float16)wb;
    } else {
        WuT[t] = (k < C) ? (_Float16)W[(size_t)k * Cout + n] : (_Float16)0.f;
    }
}

// f32 [M x ld] cols [col0,col0+C) -> f16 [M x Kpad], zero-padded K
__global__ __launch_bounds__(256) void cvt_x(const float* __restrict__ src,
                                             int ld, int col0, int C, int Kpad,
                                             _Float16* __restrict__ dst) {
    int t = blockIdx.x * blockDim.x + threadIdx.x;
    if (t >= MTOT * Kpad) return;
    int m = t / Kpad, k = t % Kpad;
    dst[t] = (k < C) ? (_Float16)src[(size_t)m * ld + col0 + k] : (_Float16)0.f;
}

// ---------------------------------------------------------------------------
// GEMM: Y[M x Cout] = act(X[M x Kpad](f16) * W(f16, stored [Cout x Kpad]) + bias)
// One wave per 16x16 output tile; v_wmma_f32_16x16x32_f16 over K.
// A layout (16-bit 16x32): lane<16 -> K {0..7,16..23}; lane>=16 -> K {8..15,24..31}.
// B layout (32x16):        lane<16 -> K {0..7,16..23} of col lane; lane>=16 -> K {8..15,24..31}.
// C/D: element e -> row e (+8 if lane>=16), col = lane&15.
// ---------------------------------------------------------------------------
template<bool RELU, bool OUT16>
__global__ __launch_bounds__(256) void gemm_wmma(const _Float16* __restrict__ X,
                                                 const _Float16* __restrict__ Wt,
                                                 const float* __restrict__ bias,
                                                 void* __restrict__ Y,
                                                 int Kpad, int ldy, int col0,
                                                 int Ntiles) {
    const int wave = (blockIdx.x * blockDim.x + threadIdx.x) >> 5;
    const int lane = threadIdx.x & 31;
    const int mt = wave / Ntiles;
    const int nt = wave % Ntiles;
    const int r16 = lane & 15;           // A row / B col within tile
    const bool hi = lane >= 16;

    const _Float16* xp = X  + (size_t)(mt * 16 + r16) * Kpad + (hi ? 8 : 0);
    const _Float16* wp = Wt + (size_t)(nt * 16 + r16) * Kpad + (hi ? 8 : 0);

    v8f acc = {};
    for (int k0 = 0; k0 < Kpad; k0 += 32) {
        h8 a0 = *(const h8*)(xp + k0);
        h8 a1 = *(const h8*)(xp + k0 + 16);
        h8 b0 = *(const h8*)(wp + k0);
        h8 b1 = *(const h8*)(wp + k0 + 16);
        v16h A, Bm;
        for (int e = 0; e < 8; ++e) {
            A[e] = a0[e];  A[e + 8] = a1[e];
            Bm[e] = b0[e]; Bm[e + 8] = b1[e];
        }
        acc = __builtin_amdgcn_wmma_f32_16x16x32_f16(false, A, false, Bm,
                                                     (short)0, acc, false, false);
    }

    const int ccol = nt * 16 + r16;
    const float bv = bias ? bias[ccol] : 0.f;
    for (int e = 0; e < 8; ++e) {
        float t = acc[e] + bv;
        if (RELU) t = t > 0.f ? t : 0.f;
        const size_t r = (size_t)(mt * 16 + e + (hi ? 8 : 0));
        if (OUT16) ((_Float16*)Y)[r * ldy + col0 + ccol] = (_Float16)t;
        else       ((float*)Y)[r * ldy + col0 + ccol] = t;
    }
}

// ---------------------------------------------------------------------------
// Edge epilogue: out[i,c] = max_{k, idx[k]!=i} relu(u[i,c] + v[idx[k],c])
// ---------------------------------------------------------------------------
template<bool OUT16>
__global__ __launch_bounds__(256) void edge_max(const _Float16* __restrict__ Uh,
                                                const _Float16* __restrict__ Vh,
                                                const int* __restrict__ idx,
                                                int Cout, void* __restrict__ Y,
                                                int ldy, int col0) {
    __shared__ int nbr[KSEL];
    const int p = blockIdx.x;
    const int i = p & (NPTS - 1);
    const int base = p - i;                 // b * NPTS
    if (threadIdx.x < KSEL) nbr[threadIdx.x] = idx[(size_t)p * KSEL + threadIdx.x];
    __syncthreads();

    for (int c = threadIdx.x; c < Cout; c += blockDim.x) {
        const float uc = (float)Uh[(size_t)p * Cout + c];
        float m = -3.0e38f;
        for (int k = 0; k < KSEL; ++k) {
            const int j = nbr[k];
            if (j == i) continue;           // self-mask (reference: -inf)
            float v = uc + (float)Vh[(size_t)(base + j) * Cout + c];
            v = v > 0.f ? v : 0.f;          // relu before max
            m = m > v ? m : v;
        }
        if (OUT16) ((_Float16*)Y)[(size_t)p * ldy + col0 + c] = (_Float16)m;
        else       ((float*)Y)[(size_t)p * ldy + col0 + c] = m;
    }
}

// Final 128 -> 1 + tanh
__global__ __launch_bounds__(256) void final_tanh(const _Float16* __restrict__ H,
                                                  const float* __restrict__ Wl3,
                                                  const float* __restrict__ bl3,
                                                  float* __restrict__ out) {
    int p = blockIdx.x * blockDim.x + threadIdx.x;
    if (p >= MTOT) return;
    float acc = bl3[0];
    const _Float16* h = H + (size_t)p * 128;
    for (int c = 0; c < 128; ++c) acc += (float)h[c] * Wl3[c];
    out[p] = tanhf(acc);
}

// ---------------------------------------------------------------------------
static inline int ceil_div(int a, int b) { return (a + b - 1) / b; }

extern "C" void kernel_launch(void* const* d_in, const int* in_sizes, int n_in,
                              void* d_out, int out_size, void* d_ws, size_t ws_size,
                              hipStream_t stream) {
    (void)in_sizes; (void)n_in; (void)out_size; (void)ws_size;
    const float* x   = (const float*)d_in[0];
    const float* W1  = (const float*)d_in[1];  const float* b1  = (const float*)d_in[2];
    const float* W2  = (const float*)d_in[3];  const float* b2  = (const float*)d_in[4];
    const float* W3  = (const float*)d_in[5];  const float* b3  = (const float*)d_in[6];
    const float* W4  = (const float*)d_in[7];  const float* b4  = (const float*)d_in[8];
    const float* Wl1 = (const float*)d_in[9];
    const float* Wl2 = (const float*)d_in[10]; const float* bl2 = (const float*)d_in[11];
    const float* Wl3 = (const float*)d_in[12]; const float* bl3 = (const float*)d_in[13];

    char* ws = (char*)d_ws;
    int*      idx  = (int*)     (ws + OFF_IDX);
    float*    xc   = (float*)   (ws + OFF_XC);   // [16384 x 208]: x1|x2|x3
    _Float16* xh   = (_Float16*)(ws + OFF_XH);
    _Float16* uh   = (_Float16*)(ws + OFF_UH);
    _Float16* vh   = (_Float16*)(ws + OFF_VH);
    _Float16* wu   = (_Float16*)(ws + OFF_WU);
    _Float16* wv   = (_Float16*)(ws + OFF_WV);
    _Float16* wl1t = (_Float16*)(ws + OFF_WL1);
    _Float16* wl2t = (_Float16*)(ws + OFF_WL2);

    // EdgeConv layers: {C_in, Kpad, Cout, knn_src_col0, knn_d, out_col0}
    // L1: x(3)   -> x1(16)  @xc[:,0:16)
    // L2: x1(16) -> x2(48)  @xc[:,16:64)
    // L3: x2(48) -> x3(144) @xc[:,64:208)
    // L4: xc(208)-> x4(512) f16 @xh, knn on x3

    // ---- Layer 1 ----
    knn_kernel<<<MTOT, 256, 0, stream>>>(x, 3, 0, 3, idx);
    prep_weight<<<ceil_div(16 * 32, 256), 256, 0, stream>>>(W1, 3, 16, 32, wu, wv, 1);
    cvt_x<<<ceil_div(MTOT * 32, 256), 256, 0, stream>>>(x, 3, 0, 3, 32, xh);
    gemm_wmma<false, true><<<128 * 1, 256, 0, stream>>>(xh, wu, b1, uh, 32, 16, 0, 1);
    gemm_wmma<false, true><<<128 * 1, 256, 0, stream>>>(xh, wv, nullptr, vh, 32, 16, 0, 1);
    edge_max<false><<<MTOT, 256, 0, stream>>>(uh, vh, idx, 16, xc, 208, 0);

    // ---- Layer 2 ----
    knn_kernel<<<MTOT, 256, 0, stream>>>(xc, 208, 0, 16, idx);
    prep_weight<<<ceil_div(48 * 32, 256), 256, 0, stream>>>(W2, 16, 48, 32, wu, wv, 1);
    cvt_x<<<ceil_div(MTOT * 32, 256), 256, 0, stream>>>(xc, 208, 0, 16, 32, xh);
    gemm_wmma<false, true><<<128 * 3, 256, 0, stream>>>(xh, wu, b2, uh, 32, 48, 0, 3);
    gemm_wmma<false, true><<<128 * 3, 256, 0, stream>>>(xh, wv, nullptr, vh, 32, 48, 0, 3);
    edge_max<false><<<MTOT, 256, 0, stream>>>(uh, vh, idx, 48, xc, 208, 16);

    // ---- Layer 3 ----
    knn_kernel<<<MTOT, 256, 0, stream>>>(xc, 208, 16, 48, idx);
    prep_weight<<<ceil_div(144 * 64, 256), 256, 0, stream>>>(W3, 48, 144, 64, wu, wv, 1);
    cvt_x<<<ceil_div(MTOT * 64, 256), 256, 0, stream>>>(xc, 208, 16, 48, 64, xh);
    gemm_wmma<false, true><<<128 * 9, 256, 0, stream>>>(xh, wu, b3, uh, 64, 144, 0, 9);
    gemm_wmma<false, true><<<128 * 9, 256, 0, stream>>>(xh, wv, nullptr, vh, 64, 144, 0, 9);
    edge_max<false><<<MTOT, 256, 0, stream>>>(uh, vh, idx, 144, xc, 208, 64);

    // ---- Layer 4 (knn on x3, features = concat x1|x2|x3) ----
    knn_kernel<<<MTOT, 256, 0, stream>>>(xc, 208, 64, 144, idx);
    prep_weight<<<ceil_div(512 * 224, 256), 256, 0, stream>>>(W4, 208, 512, 224, wu, wv, 1);
    cvt_x<<<ceil_div(MTOT * 224, 256), 256, 0, stream>>>(xc, 208, 0, 208, 224, xh);
    gemm_wmma<false, true><<<128 * 32, 256, 0, stream>>>(xh, wu, b4, uh, 224, 512, 0, 32);
    gemm_wmma<false, true><<<128 * 32, 256, 0, stream>>>(xh, wv, nullptr, vh, 224, 512, 0, 32);
    edge_max<true><<<MTOT, 256, 0, stream>>>(uh, vh, idx, 512, xh, 512, 0); // x4 (f16) -> xh

    // ---- MLP head ----
    prep_weight<<<ceil_div(256 * 512, 256), 256, 0, stream>>>(Wl1, 512, 256, 512, wl1t, nullptr, 0);
    gemm_wmma<true, true><<<128 * 16, 256, 0, stream>>>(xh, wl1t, nullptr, uh, 512, 256, 0, 16);
    prep_weight<<<ceil_div(128 * 256, 256), 256, 0, stream>>>(Wl2, 256, 128, 256, wl2t, nullptr, 0);
    gemm_wmma<true, true><<<128 * 8, 256, 0, stream>>>(uh, wl2t, bl2, vh, 256, 128, 0, 8);
    final_tanh<<<ceil_div(MTOT, 256), 256, 0, stream>>>(vh, Wl3, bl3, (float*)d_out);
}
